// Affinity_20761871909535
// MI455X (gfx1250) — matrix operand order repaired and checked by reading, not compiled, not run on hardware
//
#include <hip/hip_runtime.h>

#define HDIM 128
#define NPROT 512
#define NDRUG 64
#define BATCH 8
#define NPAIR (NPROT * NDRUG)                       /* 32768 */
#define TILES_PER_BATCH (NPAIR / 16)                /* 2048  */
#define TILES_PER_BLOCK 16
#define BLOCKS_PER_BATCH (TILES_PER_BATCH / TILES_PER_BLOCK) /* 128 */

typedef __attribute__((ext_vector_type(16))) __bf16 v16bf;
typedef __attribute__((ext_vector_type(8)))  __bf16 v8bf;
typedef __attribute__((ext_vector_type(8)))  float  v8f;

#if __has_builtin(__builtin_amdgcn_sched_barrier)
#define SCHED_FENCE() __builtin_amdgcn_sched_barrier(0)
#else
#define SCHED_FENCE()
#endif

__device__ __forceinline__ float lrelu(float x, float s) { return x >= 0.f ? x : s * x; }

// sigma: swap bits 3 and 4 of a 7-bit feature index (logical <-> physical slot)
__device__ __host__ __forceinline__ int sigma_f(int m) {
  return (m & 0x67) | ((m & 8) << 1) | ((m & 16) >> 1);
}

// sum across each 16-lane row via DPP (no LDS traffic)
__device__ __forceinline__ float rowsum16(float v) {
#if __has_builtin(__builtin_amdgcn_mov_dpp)
  int x;
  x = __builtin_amdgcn_mov_dpp(__float_as_int(v), 0xB1, 0xF, 0xF, true);  // quad_perm xor1
  v += __int_as_float(x);
  x = __builtin_amdgcn_mov_dpp(__float_as_int(v), 0x4E, 0xF, 0xF, true);  // quad_perm xor2
  v += __int_as_float(x);
  x = __builtin_amdgcn_mov_dpp(__float_as_int(v), 0x141, 0xF, 0xF, true); // row_half_mirror
  v += __int_as_float(x);
  x = __builtin_amdgcn_mov_dpp(__float_as_int(v), 0x140, 0xF, 0xF, true); // row_mirror
  v += __int_as_float(x);
#else
  v += __shfl_xor(v, 1, 32);
  v += __shfl_xor(v, 2, 32);
  v += __shfl_xor(v, 4, 32);
  v += __shfl_xor(v, 8, 32);
#endif
  return v;
}

// ---------------------------------------------------------------------------
// k_prep: swizzle 8 weight matrices into bf16 WMMA A-fragment layout + zero bufs
//   mats: 0-2 Wv[i], 3 W1a, 4 p_w, 5 d_w, 6 Wv0_top, 7 Wv0_bot
//   fragment id  = (mat*4 + kb)*8 + mb ; 1KB per fragment (32 lanes x 32B)
//   physical k   = kb*32 + 16h + j   (h = lane>>4)
//   output col   = sigma(16*mb + (lane&15))             (bit3<->bit4 swap)
//   16B units within a fragment XOR-swizzled by ((lane>>3)&1) for LDS banks
// ---------------------------------------------------------------------------
__global__ __launch_bounds__(256) void k_prep(const float* __restrict__ Wv_w,
                                              const float* __restrict__ W1a_w,
                                              const float* __restrict__ p_w,
                                              const float* __restrict__ d_w,
                                              const float* __restrict__ Wv0_w,
                                              __bf16* __restrict__ swz,
                                              float* __restrict__ zbuf) {
  const int idx = blockIdx.x * 256 + threadIdx.x;   // 131072 total
  if (idx < 3072) zbuf[idx] = 0.f;                  // u_d, u_p, s accumulators
  const int j    = idx & 15;
  const int lane = (idx >> 4) & 31;
  const int mb   = (idx >> 9) & 7;
  const int kb   = (idx >> 12) & 3;
  const int mat  = (idx >> 14) & 7;
  const int hh   = lane >> 4;
  const int k = kb * 32 + 16 * hh + j;
  const int m = sigma_f(mb * 16 + (lane & 15));
  const float* W;
  if (mat < 3)       W = Wv_w + (size_t)mat * HDIM * HDIM;
  else if (mat == 3) W = W1a_w;
  else if (mat == 4) W = p_w;
  else if (mat == 5) W = d_w;
  else if (mat == 6) W = Wv0_w;
  else               W = Wv0_w + (size_t)HDIM * HDIM;
  const int frag = ((mat * 4 + kb) * 8 + mb);
  const int unit = (2 * lane + (j >> 3)) ^ ((lane >> 3) & 1);
  swz[frag * 512 + unit * 8 + (j & 7)] = (__bf16)W[k * HDIM + m];
}

// ---------------------------------------------------------------------------
// gemm128: y[row, f] = sum_k x[row, k] * Wmat[k, f] + bias[f]   (logical f, k)
// ---------------------------------------------------------------------------
__device__ __forceinline__ void gemm128(const __bf16* lds_w, int mat,
                                        const float xin[8][8], float yout[8][8],
                                        const float* lds_bias, int bias_off,
                                        int lane, int h) {
  v8f acc[8] = {};
  const int sw = (lane >> 3) & 1;
  const int u0 = ((2 * lane + 0) ^ sw) * 8;  // bf16 offset of low 16B
  const int u1 = ((2 * lane + 1) ^ sw) * 8;  // bf16 offset of high 16B
#pragma unroll
  for (int kb = 0; kb < 4; ++kb) {
    v16bf bf;
#pragma unroll
    for (int v = 0; v < 8; ++v) {
      bf[v]     = (__bf16)xin[2 * kb][v];
      bf[v + 8] = (__bf16)xin[2 * kb + 1][v];
    }
    const __bf16* wb = lds_w + (size_t)((mat * 4 + kb) * 8) * 512;
#pragma unroll
    for (int mb = 0; mb < 8; ++mb) {
      const __bf16* fb = wb + mb * 512;
      v8bf lo = *(const v8bf*)(fb + u0);
      v8bf hi = *(const v8bf*)(fb + u1);
      v16bf af = __builtin_shufflevector(lo, hi, 0, 1, 2, 3, 4, 5, 6, 7,
                                         8, 9, 10, 11, 12, 13, 14, 15);
      acc[mb] = __builtin_amdgcn_wmma_f32_16x16x32_bf16(
          false, af, false, bf, (short)0, acc[mb], false, false);
    }
  }
#pragma unroll
  for (int mb = 0; mb < 8; ++mb) {
    const int f0 = 32 * (mb >> 1) + 16 * h + 8 * (mb & 1);  // sigma(16mb+8h)
#pragma unroll
    for (int v = 0; v < 8; ++v) yout[mb][v] = acc[mb][v] + lds_bias[bias_off + f0 + v];
  }
}

// ---------------------------------------------------------------------------
// k_projw: WMMA projection stage.
//   feat = lrelu(X@W + b, .1)*mask ; u += sum_rows(feat) ; OUT = feat @ Wv0half
//   blocks 0..31 prot (mats {4,6}), blocks 32..35 drug (mats {5,7}).
// ---------------------------------------------------------------------------
__global__ __launch_bounds__(256) void k_projw(
    const float* __restrict__ prot, const float* __restrict__ drug,
    const float* __restrict__ pu_mask, const float* __restrict__ du_mask,
    const float* __restrict__ p_b, const float* __restrict__ d_b,
    const __bf16* __restrict__ swz,
    float* __restrict__ PP, float* __restrict__ DP,
    float* __restrict__ u_p, float* __restrict__ u_d) {
  __shared__ __align__(32) __bf16 lds_w[2 * 4 * 8 * 512];  // 64 KB (2 mats)
  __shared__ float lds_bias[256];  // [0,128)=bias  [128,256)=zeros
  const bool isProt = blockIdx.x < 32;
  const int matA = isProt ? 4 : 5;   // p_w / d_w
  const int matB = isProt ? 6 : 7;   // Wv0 top / bottom
  {
    const uint4* srcA = (const uint4*)(swz + (size_t)matA * 16384);
    const uint4* srcB = (const uint4*)(swz + (size_t)matB * 16384);
    uint4* dst = (uint4*)lds_w;
#pragma unroll
    for (int i = 0; i < 8; ++i) dst[threadIdx.x + i * 256] = srcA[threadIdx.x + i * 256];
#pragma unroll
    for (int i = 0; i < 8; ++i) dst[2048 + threadIdx.x + i * 256] = srcB[threadIdx.x + i * 256];
    if (threadIdx.x < 128) lds_bias[threadIdx.x] = isProt ? p_b[threadIdx.x] : d_b[threadIdx.x];
    else                   lds_bias[threadIdx.x] = 0.f;
  }
  __syncthreads();

  const int wave = threadIdx.x >> 5;
  const int lane = threadIdx.x & 31;
  const int h    = lane >> 4;
  const int nrow = lane & 15;
  const int tile = (isProt ? blockIdx.x : (blockIdx.x - 32)) * 8 + wave;
  const int r = tile * 16 + nrow;                  // row within region
  const int b = isProt ? (r / NPROT) : (r / NDRUG);
  const float* Xrow = (isProt ? prot : drug) + (size_t)r * HDIM;
  const float mval = isProt ? pu_mask[r] : du_mask[r];
  float* OUT = (isProt ? PP : DP) + (size_t)r * HDIM;
  float* U   = (isProt ? u_p : u_d) + b * HDIM;

  float x[8][8], y[8][8];
#pragma unroll
  for (int mb = 0; mb < 8; ++mb) {
    const int f0 = 32 * (mb >> 1) + 16 * h + 8 * (mb & 1);
    const float4 a0 = *(const float4*)(Xrow + f0);
    const float4 a1 = *(const float4*)(Xrow + f0 + 4);
    x[mb][0] = a0.x; x[mb][1] = a0.y; x[mb][2] = a0.z; x[mb][3] = a0.w;
    x[mb][4] = a1.x; x[mb][5] = a1.y; x[mb][6] = a1.z; x[mb][7] = a1.w;
  }
  SCHED_FENCE();
  gemm128(lds_w, 0, x, y, lds_bias, 0, lane, h);   // X@W + bias
  SCHED_FENCE();
#pragma unroll
  for (int mb = 0; mb < 8; ++mb)
#pragma unroll
    for (int v = 0; v < 8; ++v) y[mb][v] = lrelu(y[mb][v], 0.1f) * mval;
  // u += sum over rows of feat
#pragma unroll
  for (int mb = 0; mb < 8; ++mb) {
    const int f0 = 32 * (mb >> 1) + 16 * h + 8 * (mb & 1);
#pragma unroll
    for (int v = 0; v < 8; ++v) {
      const float val = rowsum16(y[mb][v]);
      if (nrow == v) atomicAdd(&U[f0 + v], val);
    }
  }
  SCHED_FENCE();
  gemm128(lds_w, 1, y, x, lds_bias, 128, lane, h);  // feat @ Wv0half (zero bias)
  SCHED_FENCE();
#pragma unroll
  for (int mb = 0; mb < 8; ++mb) {
    const int f0 = 32 * (mb >> 1) + 16 * h + 8 * (mb & 1);
    float4 o0 = {x[mb][0], x[mb][1], x[mb][2], x[mb][3]};
    float4 o1 = {x[mb][4], x[mb][5], x[mb][6], x[mb][7]};
    *(float4*)(OUT + f0)     = o0;
    *(float4*)(OUT + f0 + 4) = o1;
  }
}

// ---------------------------------------------------------------------------
// k_main: fused pair-tensor pipeline. One wave = 16 (p,d) pair rows.
// ---------------------------------------------------------------------------
__global__ __launch_bounds__(256) void k_main(
    const float* __restrict__ PP, const float* __restrict__ DP,
    const __bf16* __restrict__ swz,
    const float* __restrict__ Wv0_b, const float* __restrict__ Wv_b,
    const float* __restrict__ att_w, const float* __restrict__ att_b,
    const float* __restrict__ W1a_b, float* __restrict__ s_out) {
  __shared__ __align__(32) __bf16 lds_w[4 * 4 * 8 * 512];  // 128 KB (mats 0-3)
  __shared__ float lds_bias[1024];  // [0,128)=Wv0_b [128,512)=Wv_b [512,896)=att_w [896,1024)=W1a_b
  __shared__ float lds_s[HDIM];

  {  // cooperative LDS fill (global layout == LDS layout; mats 0-3 are first)
    const uint4* src = (const uint4*)swz;
    uint4* dst = (uint4*)lds_w;
#pragma unroll
    for (int i = 0; i < 32; ++i) dst[threadIdx.x + i * 256] = src[threadIdx.x + i * 256];
    for (int i = threadIdx.x; i < 1024; i += 256) {
      float v;
      if (i < 128)      v = Wv0_b[i];
      else if (i < 512) v = Wv_b[i - 128];
      else if (i < 896) v = att_w[i - 512];
      else              v = W1a_b[i - 896];
      lds_bias[i] = v;
    }
    if (threadIdx.x < HDIM) lds_s[threadIdx.x] = 0.f;
  }
  __syncthreads();

  const int b    = blockIdx.x / BLOCKS_PER_BATCH;
  const int blk  = blockIdx.x % BLOCKS_PER_BATCH;
  const int wave = threadIdx.x >> 5;
  const int lane = threadIdx.x & 31;
  const int h    = lane >> 4;
  const int nrow = lane & 15;
  const float abv[3] = {att_b[0], att_b[1], att_b[2]};

#pragma unroll 1
  for (int t = 0; t < TILES_PER_BLOCK / 8; ++t) {
    const int tile = blk * TILES_PER_BLOCK + t * 8 + wave;
    const int pr = tile * 16 + nrow;        // pair index within batch
    const int p = pr >> 6;                  // / ND
    const int d = pr & 63;                  // % ND
    const float* pp = PP + ((size_t)b * NPROT + p) * HDIM;
    const float* dp = DP + ((size_t)b * NDRUG + d) * HDIM;

    float x[8][8], y[8][8];
    // x0 = lrelu(PP[p] + DP[d] + Wv0_b, 0.1)  (slot (mb,v) holds logical f0+v)
#pragma unroll
    for (int mb = 0; mb < 8; ++mb) {
      const int f0 = 32 * (mb >> 1) + 16 * h + 8 * (mb & 1);
      const float4 pa  = *(const float4*)(pp + f0);
      const float4 pb4 = *(const float4*)(pp + f0 + 4);
      const float4 da  = *(const float4*)(dp + f0);
      const float4 db4 = *(const float4*)(dp + f0 + 4);
      const float pv[8] = {pa.x, pa.y, pa.z, pa.w, pb4.x, pb4.y, pb4.z, pb4.w};
      const float dv[8] = {da.x, da.y, da.z, da.w, db4.x, db4.y, db4.z, db4.w};
#pragma unroll
      for (int v = 0; v < 8; ++v)
        x[mb][v] = lrelu(pv[v] + dv[v] + lds_bias[f0 + v], 0.1f);
    }
    SCHED_FENCE();

    // depth loop: y = x@Wv[i]+b ; alpha = y.att[i]+ab ; x += alpha*y
#pragma unroll
    for (int i = 0; i < 3; ++i) {
      gemm128(lds_w, i, x, y, lds_bias, 128 + i * 128, lane, h);
      SCHED_FENCE();
      float part = 0.f;
#pragma unroll
      for (int mb = 0; mb < 8; ++mb) {
        const int f0 = 32 * (mb >> 1) + 16 * h + 8 * (mb & 1);
#pragma unroll
        for (int v = 0; v < 8; ++v) part += y[mb][v] * lds_bias[512 + i * 128 + f0 + v];
      }
      const float alpha = part + __shfl_xor(part, 16, 32) + abv[i];
      if (i < 2) {
#pragma unroll
        for (int mb = 0; mb < 8; ++mb)
#pragma unroll
          for (int v = 0; v < 8; ++v) x[mb][v] += alpha * y[mb][v];
      } else {
#pragma unroll
        for (int mb = 0; mb < 8; ++mb)
#pragma unroll
          for (int v = 0; v < 8; ++v) {
            const float gg = alpha * y[mb][v];
            x[mb][v] += gg;              // mij_final
            y[mb][v] = x[mb][v] + gg;    // h_fin = g + mij_final
          }
      }
      SCHED_FENCE();
    }

    // t = relu(h_fin@W1a + b) ; s += sum_rows(t)   (output overwrites x)
    gemm128(lds_w, 3, y, x, lds_bias, 896, lane, h);
    SCHED_FENCE();
#pragma unroll
    for (int mb = 0; mb < 8; ++mb) {
      const int f0 = 32 * (mb >> 1) + 16 * h + 8 * (mb & 1);
#pragma unroll
      for (int v = 0; v < 8; ++v) {
        const float val = rowsum16(fmaxf(x[mb][v], 0.f));  // sum over 16 rows
        if (nrow == v) atomicAdd(&lds_s[f0 + v], val);
      }
    }
    SCHED_FENCE();
  }
  __syncthreads();
  if (threadIdx.x < HDIM) atomicAdd(&s_out[b * HDIM + threadIdx.x], lds_s[threadIdx.x]);
}

// ---------------------------------------------------------------------------
// k_final: tiny per-batch MLP tail. global1 = s@W1b + NPAIR*b1b.
// ---------------------------------------------------------------------------
__global__ __launch_bounds__(256) void k_final(
    const float* __restrict__ u_d, const float* __restrict__ u_p,
    const float* __restrict__ s,
    const float* __restrict__ Wu_w, const float* __restrict__ Wu_b,
    const float* __restrict__ W1b_w, const float* __restrict__ W1b_b,
    const float* __restrict__ W2a_w, const float* __restrict__ W2a_b,
    const float* __restrict__ W2b_w, const float* __restrict__ W2b_b,
    const float* __restrict__ W3_w, const float* __restrict__ W3_b,
    const float* __restrict__ Wu1a_w, const float* __restrict__ Wu1a_b,
    const float* __restrict__ Wu1b_w, const float* __restrict__ Wu1b_b,
    const float* __restrict__ W5_w, const float* __restrict__ W5_b,
    float* __restrict__ out) {
  const int b = blockIdx.x;
  const int j = threadIdx.x;
  __shared__ float m_u[HDIM], g1[HDIM], h2[2 * HDIM], g2[HDIM], g3[HDIM], t1[HDIM], mu2[HDIM];
  if (j < HDIM) {
    float a = Wu_b[j];
    for (int k = 0; k < HDIM; ++k) a += u_d[b * HDIM + k] * Wu_w[k * HDIM + j];
    for (int k = 0; k < HDIM; ++k) a += u_p[b * HDIM + k] * Wu_w[(HDIM + k) * HDIM + j];
    m_u[j] = lrelu(a, 0.01f);
    float gg = (float)NPAIR * W1b_b[j];
    for (int k = 0; k < HDIM; ++k) gg += s[b * HDIM + k] * W1b_w[k * HDIM + j];
    g1[j] = gg;
  }
  __syncthreads();
  {
    float a = W2a_b[j];
    for (int k = 0; k < HDIM; ++k) a += g1[k] * W2a_w[k * 2 * HDIM + j];
    h2[j] = lrelu(a, 0.1f);
  }
  __syncthreads();
  if (j < HDIM) {
    float a = W2b_b[j];
    for (int k = 0; k < 2 * HDIM; ++k) a += h2[k] * W2b_w[k * HDIM + j];
    g2[j] = a;
  }
  __syncthreads();
  if (j < HDIM) {
    float a = W3_b[j];
    for (int k = 0; k < HDIM; ++k) a += g2[k] * W3_w[k * HDIM + j];
    g3[j] = lrelu(a, 0.1f);
  }
  __syncthreads();
  if (j < HDIM) {
    float a = Wu1a_b[j];
    for (int k = 0; k < HDIM; ++k) a += m_u[k] * Wu1a_w[k * HDIM + j];
    for (int k = 0; k < HDIM; ++k) a += g3[k] * Wu1a_w[(HDIM + k) * HDIM + j];
    t1[j] = lrelu(a, 0.1f);
  }
  __syncthreads();
  if (j < HDIM) {
    float a = Wu1b_b[j];
    for (int k = 0; k < HDIM; ++k) a += t1[k] * Wu1b_w[k * HDIM + j];
    mu2[j] = a;
  }
  __syncthreads();
  if (j == 0) {
    float a = W5_b[0];
    for (int k = 0; k < HDIM; ++k) a += mu2[k] * W5_w[k];
    out[b] = a;
  }
}

// ---------------------------------------------------------------------------
extern "C" void kernel_launch(void* const* d_in, const int* in_sizes, int n_in,
                              void* d_out, int out_size, void* d_ws, size_t ws_size,
                              hipStream_t stream) {
  const float* prot    = (const float*)d_in[0];
  const float* drug    = (const float*)d_in[1];
  const float* pu_mask = (const float*)d_in[2];
  const float* du_mask = (const float*)d_in[3];
  const float* d_w = (const float*)d_in[4];
  const float* d_b = (const float*)d_in[5];
  const float* p_w = (const float*)d_in[6];
  const float* p_b = (const float*)d_in[7];
  const float* Wv0_w = (const float*)d_in[8];
  const float* Wv0_b = (const float*)d_in[9];
  const float* att_w = (const float*)d_in[10];
  const float* att_b = (const float*)d_in[11];
  const float* Wv_w  = (const float*)d_in[12];
  const float* Wv_b  = (const float*)d_in[13];
  const float* Wu_w  = (const float*)d_in[14];
  const float* Wu_b  = (const float*)d_in[15];
  const float* W1a_w = (const float*)d_in[16];
  const float* W1a_b = (const float*)d_in[17];
  const float* W1b_w = (const float*)d_in[18];
  const float* W1b_b = (const float*)d_in[19];
  const float* W2a_w = (const float*)d_in[20];
  const float* W2a_b = (const float*)d_in[21];
  const float* W2b_w = (const float*)d_in[22];
  const float* W2b_b = (const float*)d_in[23];
  const float* W3_w  = (const float*)d_in[24];
  const float* W3_b  = (const float*)d_in[25];
  const float* Wu1a_w = (const float*)d_in[26];
  const float* Wu1a_b = (const float*)d_in[27];
  const float* Wu1b_w = (const float*)d_in[28];
  const float* Wu1b_b = (const float*)d_in[29];
  const float* W5_w  = (const float*)d_in[30];
  const float* W5_b  = (const float*)d_in[31];

  float* ws  = (float*)d_ws;
  float* PP  = ws;                 // 8*512*128 = 524288 f32
  float* DP  = ws + 524288;        // 8*64*128  = 65536 f32
  float* u_d = ws + 589824;        // 1024
  float* u_p = ws + 590848;        // 1024
  float* s   = ws + 591872;        // 1024
  __bf16* swz = (__bf16*)(ws + 592896);  // 131072 bf16 swizzled weights (8 mats)

  k_prep<<<512, 256, 0, stream>>>(Wv_w, W1a_w, p_w, d_w, Wv0_w, swz, u_d);
  k_projw<<<36, 256, 0, stream>>>(prot, drug, pu_mask, du_mask, p_b, d_b, swz,
                                  PP, DP, u_p, u_d);
  k_main<<<BATCH * BLOCKS_PER_BATCH, 256, 0, stream>>>(
      PP, DP, swz, Wv0_b, Wv_b, att_w, att_b, W1a_b, s);
  k_final<<<BATCH, 256, 0, stream>>>(
      u_d, u_p, s, Wu_w, Wu_b, W1b_w, W1b_b, W2a_w, W2a_b, W2b_w, W2b_b,
      W3_w, W3_b, Wu1a_w, Wu1a_b, Wu1b_w, Wu1b_b, W5_w, W5_b, (float*)d_out);
}
